// CausalSelfAttention_51496657879127
// MI455X (gfx1250) — compile-verified, matrix operand order checked by reading, not run on hardware
//
#include <hip/hip_runtime.h>
#include <hip/hip_bf16.h>

// ---------------- types ----------------
typedef __bf16 bf16x8  __attribute__((ext_vector_type(8)));
typedef __bf16 bf16x16 __attribute__((ext_vector_type(16)));
typedef float  f32x8   __attribute__((ext_vector_type(8)));

#define Bb 2
#define Tt 2048
#define Cc 1024
#define Hh 16
#define Dd 64
#define BT (Bb*Tt)      // 4096

// WMMA bf16 16x16x32, fp32 accum
__device__ __forceinline__ f32x8 wmma_bf16(bf16x16 a, bf16x16 b, f32x8 c) {
  return __builtin_amdgcn_wmma_f32_16x16x32_bf16(false, a, false, b, (short)0, c,
                                                 false, false);
}

// Load a 16-lane fragment (A-row or B-column == row of the K-contiguous source).
// CDNA5 16-bit A layout: lanes 0-15 hold K = {k0..k0+7, k0+16..k0+23},
// lanes 16-31 hold K = {k0+8..k0+15, k0+24..k0+31}.  -> two b128 loads.
__device__ __forceinline__ bf16x16 load_frag(const __bf16* __restrict__ base,
                                             int ld, int row, int k0, int half) {
  const __bf16* p = base + row * ld + k0 + 8 * half;
  bf16x8 lo = *reinterpret_cast<const bf16x8*>(p);
  bf16x8 hi = *reinterpret_cast<const bf16x8*>(p + 16);
  union { bf16x16 v; bf16x8 h[2]; } u;
  u.h[0] = lo; u.h[1] = hi;
  return u.v;
}

// ---------------- kernel 1: fp32 -> bf16 pack ----------------
__global__ __launch_bounds__(256) void pack_kernel(
    const float* __restrict__ x, const float* __restrict__ w1,
    const float* __restrict__ w2, __bf16* __restrict__ xb,
    __bf16* __restrict__ w1b, __bf16* __restrict__ w2b) {
  int stride = gridDim.x * blockDim.x;
  int i0 = blockIdx.x * blockDim.x + threadIdx.x;
  for (int i = i0; i < BT * Cc;     i += stride) xb[i]  = (__bf16)x[i];
  for (int i = i0; i < 3 * Cc * Cc; i += stride) w1b[i] = (__bf16)w1[i];
  for (int i = i0; i < Cc * Cc;     i += stride) w2b[i] = (__bf16)w2[i];
}

// -------- shared GEMM core: wave computes 32x64 tile, double-buffered --------
// C[m][n] = sum_k A[m][k] * B[n][k]   (B is "weights": columns of B^T are rows)
__device__ __forceinline__ void gemm_core_32x64(
    const __bf16* __restrict__ A, const __bf16* __restrict__ B,
    int m0, int n0, int lsub, int half, f32x8 acc[2][4]) {
  // preload K-step 0
  bf16x16 a0 = load_frag(A, Cc, m0 + lsub,      0, half);
  bf16x16 a1 = load_frag(A, Cc, m0 + 16 + lsub, 0, half);
  bf16x16 b0 = load_frag(B, Cc, n0 + 0  + lsub, 0, half);
  bf16x16 b1 = load_frag(B, Cc, n0 + 16 + lsub, 0, half);
  bf16x16 b2 = load_frag(B, Cc, n0 + 32 + lsub, 0, half);
  bf16x16 b3 = load_frag(B, Cc, n0 + 48 + lsub, 0, half);
  for (int kt = 0; kt < Cc / 32 - 1; ++kt) {
    const int k0 = (kt + 1) * 32;
    // issue next K-step's loads BEFORE consuming current frags
    bf16x16 na0 = load_frag(A, Cc, m0 + lsub,      k0, half);
    bf16x16 na1 = load_frag(A, Cc, m0 + 16 + lsub, k0, half);
    bf16x16 nb0 = load_frag(B, Cc, n0 + 0  + lsub, k0, half);
    bf16x16 nb1 = load_frag(B, Cc, n0 + 16 + lsub, k0, half);
    bf16x16 nb2 = load_frag(B, Cc, n0 + 32 + lsub, k0, half);
    bf16x16 nb3 = load_frag(B, Cc, n0 + 48 + lsub, k0, half);
    acc[0][0] = wmma_bf16(a0, b0, acc[0][0]);
    acc[0][1] = wmma_bf16(a0, b1, acc[0][1]);
    acc[0][2] = wmma_bf16(a0, b2, acc[0][2]);
    acc[0][3] = wmma_bf16(a0, b3, acc[0][3]);
    acc[1][0] = wmma_bf16(a1, b0, acc[1][0]);
    acc[1][1] = wmma_bf16(a1, b1, acc[1][1]);
    acc[1][2] = wmma_bf16(a1, b2, acc[1][2]);
    acc[1][3] = wmma_bf16(a1, b3, acc[1][3]);
    a0 = na0; a1 = na1; b0 = nb0; b1 = nb1; b2 = nb2; b3 = nb3;
  }
  acc[0][0] = wmma_bf16(a0, b0, acc[0][0]);
  acc[0][1] = wmma_bf16(a0, b1, acc[0][1]);
  acc[0][2] = wmma_bf16(a0, b2, acc[0][2]);
  acc[0][3] = wmma_bf16(a0, b3, acc[0][3]);
  acc[1][0] = wmma_bf16(a1, b0, acc[1][0]);
  acc[1][1] = wmma_bf16(a1, b1, acc[1][1]);
  acc[1][2] = wmma_bf16(a1, b2, acc[1][2]);
  acc[1][3] = wmma_bf16(a1, b3, acc[1][3]);
}

// ---------------- kernel 2: QKV GEMM + scatter ----------------
// M=4096, N=3072, K=1024; block = 8 waves; wave 32x64; block tile 128x128
__global__ __launch_bounds__(256) void qkv_kernel(
    const __bf16* __restrict__ X, const __bf16* __restrict__ W,
    __bf16* __restrict__ Qb, __bf16* __restrict__ Kb, __bf16* __restrict__ VTb) {
  const int lane = threadIdx.x & 31;
  const int wid  = threadIdx.x >> 5;
  const int half = lane >> 4;
  const int lsub = lane & 15;
  const int m0 = blockIdx.y * 128 + (wid >> 1) * 32;
  const int n0 = blockIdx.x * 128 + (wid & 1) * 64;

  f32x8 acc[2][4] = {};
  gemm_core_32x64(X, W, m0, n0, lsub, half, acc);

  // scatter: f<C -> Q[B,H,T,D]; f<2C -> K[B,H,T,D]; else -> VT[B,H,D,T]
#pragma unroll
  for (int a2 = 0; a2 < 2; ++a2)
#pragma unroll
    for (int nt = 0; nt < 4; ++nt) {
      int fb  = n0 + nt * 16;
      int seg = fb >> 10;            // 0:Q 1:K 2:V
      int cf  = fb & (Cc - 1);
      int h   = cf >> 6;
      int d   = (cf & 63) + lsub;
#pragma unroll
      for (int r = 0; r < 8; ++r) {
        int m  = m0 + a2 * 16 + r + 8 * half;
        int b_ = m >> 11, t = m & (Tt - 1);
        __bf16 v = (__bf16)acc[a2][nt][r];
        if (seg == 0)      Qb [(((b_ * Hh + h) * Tt) + t) * Dd + d] = v;
        else if (seg == 1) Kb [(((b_ * Hh + h) * Tt) + t) * Dd + d] = v;
        else               VTb[(((b_ * Hh + h) * Dd) + d) * Tt + t] = v;
      }
    }
}

// ---------------- kernel 3: causal flash attention ----------------
// 1 wave per 16-query tile; key blocks of 32; online softmax fp32.
// Pipelined: V loads first, next-K prefetch second, so PV wmma only waits
// for V (loadcnt<=8) and QK^T wmma consumes one-iteration-old K frags.
__global__ __launch_bounds__(128) void attn_kernel(
    const __bf16* __restrict__ Qb, const __bf16* __restrict__ Kb,
    const __bf16* __restrict__ VTb, __bf16* __restrict__ A2) {
  __shared__ __attribute__((aligned(16))) __bf16 plds[4][16 * 32];
  const int lane = threadIdx.x & 31;
  const int wid  = threadIdx.x >> 5;
  const int half = lane >> 4;
  const int lsub = lane & 15;

  int g  = blockIdx.x * 4 + wid;       // global q-tile id, 0..4095
  int bh = g >> 7;                     // (B*H) index (T/16 = 128 tiles/head)
  int qt = g & 127;
  int qi = qt * 16;
  const __bf16* Qh = Qb  + (size_t)bh * Tt * Dd;
  const __bf16* Kh = Kb  + (size_t)bh * Tt * Dd;
  const __bf16* Vh = VTb + (size_t)bh * Dd * Tt;
  __bf16* lp = plds[wid];

  // Q a-frags (16 x 64 over two K=32 chunks), loaded once
  bf16x16 qa0 = load_frag(Qh, Dd, qi + lsub, 0, half);
  bf16x16 qa1 = load_frag(Qh, Dd, qi + lsub, 32, half);

  f32x8 o[4] = {};
  float mrun[8], lrun[8];
#pragma unroll
  for (int r = 0; r < 8; ++r) { mrun[r] = -1e30f; lrun[r] = 0.f; }

  const int nblk = (qt >> 1) + 1;      // only blocks with j <= qi (causal)

  // preload K b-frags for block 0
  bf16x16 kb0 = load_frag(Kh, Dd, lsub,      0,  half);
  bf16x16 kb1 = load_frag(Kh, Dd, lsub,      32, half);
  bf16x16 kb2 = load_frag(Kh, Dd, 16 + lsub, 0,  half);
  bf16x16 kb3 = load_frag(Kh, Dd, 16 + lsub, 32, half);

  for (int jb = 0; jb < nblk; ++jb) {
    int j = jb * 32;
    // (a) current-block V b-frags (needed soonest among new loads)
    bf16x16 vb0 = load_frag(Vh, Tt, 0  + lsub, j, half);
    bf16x16 vb1 = load_frag(Vh, Tt, 16 + lsub, j, half);
    bf16x16 vb2 = load_frag(Vh, Tt, 32 + lsub, j, half);
    bf16x16 vb3 = load_frag(Vh, Tt, 48 + lsub, j, half);
    // (b) next-block K b-frags (clamped, branch-free)
    int jn = (jb + 1 < nblk) ? (j + 32) : j;
    bf16x16 nk0 = load_frag(Kh, Dd, jn + lsub,      0,  half);
    bf16x16 nk1 = load_frag(Kh, Dd, jn + lsub,      32, half);
    bf16x16 nk2 = load_frag(Kh, Dd, jn + 16 + lsub, 0,  half);
    bf16x16 nk3 = load_frag(Kh, Dd, jn + 16 + lsub, 32, half);

    // (c) S = Q K^T for keys [j, j+32)
    f32x8 s0 = {}, s1 = {};
    s0 = wmma_bf16(qa0, kb0, s0);
    s0 = wmma_bf16(qa1, kb1, s0);
    s1 = wmma_bf16(qa0, kb2, s1);
    s1 = wmma_bf16(qa1, kb3, s1);

    // (d) online softmax (VALU work overlaps in-flight loads)
    const float sc = 0.03125f;         // 1/sqrt(C) = 1/32
    const bool maskIt = (jb == nblk - 1);
    float scl[8];
#pragma unroll
    for (int r = 0; r < 8; ++r) {
      float v0 = s0[r] * sc, v1 = s1[r] * sc;
      if (maskIt) {
        int qrow = qi + r + 8 * half;
        if (j + lsub > qrow)      v0 = -1e30f;
        if (j + 16 + lsub > qrow) v1 = -1e30f;
      }
      float rm = fmaxf(v0, v1);
#pragma unroll
      for (int off = 8; off >= 1; off >>= 1) rm = fmaxf(rm, __shfl_xor(rm, off, 32));
      float mn = fmaxf(mrun[r], rm);
      float sf = __expf(mrun[r] - mn);
      mrun[r]  = mn;
      float p0 = __expf(v0 - mn), p1 = __expf(v1 - mn);
      float rs = p0 + p1;
#pragma unroll
      for (int off = 8; off >= 1; off >>= 1) rs += __shfl_xor(rs, off, 32);
      lrun[r] = lrun[r] * sf + rs;
      scl[r]  = sf;
      s0[r] = p0; s1[r] = p1;
    }
#pragma unroll
    for (int nt = 0; nt < 4; ++nt)
#pragma unroll
      for (int r = 0; r < 8; ++r) o[nt][r] *= scl[r];

    // (e) P: C/D-layout -> A-layout transpose through per-wave LDS (bf16)
#pragma unroll
    for (int r = 0; r < 8; ++r) {
      int m = r + 8 * half;
      lp[m * 32 + lsub]      = (__bf16)s0[r];
      lp[m * 32 + 16 + lsub] = (__bf16)s1[r];
    }
    asm volatile("s_wait_dscnt 0" ::: "memory");
    bf16x16 pa;
    {
      const __bf16* p = &lp[lsub * 32 + 8 * half];
      union { bf16x16 v; bf16x8 h[2]; } u;
      u.h[0] = *reinterpret_cast<const bf16x8*>(p);
      u.h[1] = *reinterpret_cast<const bf16x8*>(p + 16);
      pa = u.v;
    }
    // (f) O += P * V  (only needs V loads complete; K prefetch may stay in flight)
    o[0] = wmma_bf16(pa, vb0, o[0]);
    o[1] = wmma_bf16(pa, vb1, o[1]);
    o[2] = wmma_bf16(pa, vb2, o[2]);
    o[3] = wmma_bf16(pa, vb3, o[3]);

    kb0 = nk0; kb1 = nk1; kb2 = nk2; kb3 = nk3;
  }

  // normalize and store to A2[B,T,C] (= [b][t][h*D+d]) as bf16
  int b_ = bh >> 4, h = bh & 15;
#pragma unroll
  for (int r = 0; r < 8; ++r) {
    float inv = 1.0f / lrun[r];
    int t = qi + r + 8 * half;
#pragma unroll
    for (int nt = 0; nt < 4; ++nt)
      A2[((size_t)(b_ * Tt + t)) * Cc + h * Dd + nt * 16 + lsub] =
          (__bf16)(o[nt][r] * inv);
  }
}

// ---------------- kernel 4: output projection ----------------
// out[m][n] = sum_k A2[m][k] * W2[n][k];  M=4096, N=K=1024, fp32 out
__global__ __launch_bounds__(256) void proj_kernel(
    const __bf16* __restrict__ A2, const __bf16* __restrict__ W2,
    float* __restrict__ out) {
  const int lane = threadIdx.x & 31;
  const int wid  = threadIdx.x >> 5;
  const int half = lane >> 4;
  const int lsub = lane & 15;
  const int m0 = blockIdx.y * 128 + (wid >> 1) * 32;
  const int n0 = blockIdx.x * 128 + (wid & 1) * 64;

  f32x8 acc[2][4] = {};
  gemm_core_32x64(A2, W2, m0, n0, lsub, half, acc);

#pragma unroll
  for (int a2 = 0; a2 < 2; ++a2)
#pragma unroll
    for (int nt = 0; nt < 4; ++nt)
#pragma unroll
      for (int r = 0; r < 8; ++r) {
        int m = m0 + a2 * 16 + r + 8 * half;
        out[(size_t)m * Cc + n0 + nt * 16 + lsub] = acc[a2][nt][r];
      }
}

// ---------------- launcher ----------------
extern "C" void kernel_launch(void* const* d_in, const int* in_sizes, int n_in,
                              void* d_out, int out_size, void* d_ws, size_t ws_size,
                              hipStream_t stream) {
  (void)in_sizes; (void)n_in; (void)out_size; (void)ws_size;
  const float* x      = (const float*)d_in[0];
  const float* w_attn = (const float*)d_in[1];
  const float* w_proj = (const float*)d_in[2];
  float* out = (float*)d_out;

  char* ws = (char*)d_ws;
  const size_t MB = 1024 * 1024;
  __bf16* Xb  = (__bf16*)(ws + 0);          //  8 MB: [4096][1024]
  __bf16* W1b = (__bf16*)(ws + 8  * MB);    //  6 MB: [3072][1024]
  __bf16* W2b = (__bf16*)(ws + 14 * MB);    //  2 MB: [1024][1024]
  __bf16* Qb  = (__bf16*)(ws + 16 * MB);    //  8 MB: [B,H,T,D]
  __bf16* Kb  = (__bf16*)(ws + 24 * MB);    //  8 MB: [B,H,T,D]
  __bf16* VTb = (__bf16*)(ws + 32 * MB);    //  8 MB: [B,H,D,T]
  __bf16* A2  = (__bf16*)(ws + 40 * MB);    //  8 MB: [4096][1024]

  pack_kernel<<<2048, 256, 0, stream>>>(x, w_attn, w_proj, Xb, W1b, W2b);
  qkv_kernel<<<dim3(3 * Cc / 128, BT / 128), 256, 0, stream>>>(Xb, W1b, Qb, Kb, VTb);
  attn_kernel<<<(Bb * Hh * (Tt / 16)) / 4, 128, 0, stream>>>(Qb, Kb, VTb, A2);
  proj_kernel<<<dim3(Cc / 128, BT / 128), 256, 0, stream>>>(A2, W2b, out);
}
